// WhisperAttention_59511066853784
// MI455X (gfx1250) — compile-verified, matrix operand order checked by reading, not run on hardware
//
#include <hip/hip_runtime.h>
#include <hip/hip_bf16.h>
#include <stdint.h>

// ---------------------------------------------------------------------------
// Whisper varlen attention for gfx1250 (MI455X, wave32, WMMA bf16).
// Pipeline: f32->bf16 casts -> QKV GEMM (WMMA, async-LDS double-buffered)
//           -> flash attention (WMMA, async K staging)
//           -> out GEMM (WMMA, async-LDS double-buffered).
// ---------------------------------------------------------------------------

typedef __attribute__((ext_vector_type(16))) __bf16 v16bf;
typedef __attribute__((ext_vector_type(8)))  float  v8f;

#define EMBED   1024
#define NHEADS  16
#define HDIM    64
#define QKVDIM  3072   // 3 * EMBED

union Frag16 { v16bf v; uint4 q[2]; };

__device__ __forceinline__ v8f wmma_bf16(v16bf a, v16bf b, v8f c) {
  // D(f32,16x16) = A(bf16,16x32) * B(bf16,32x16) + C
  return __builtin_amdgcn_wmma_f32_16x16x32_bf16(
      /*neg_a=*/false, a, /*neg_b=*/false, b,
      /*c_mod=*/(short)0, c, /*reuse_a=*/false, /*reuse_b=*/false);
}

// Low 32 bits of a generic pointer to __shared__ == LDS byte address
// (ISA: the aperture is checked on the high 32 bits; the low 32 bits give the LDS address).
__device__ __forceinline__ uint32_t lds_addr32(const void* p) {
  return (uint32_t)(uintptr_t)p;
}

// CDNA5 async memory->LDS copy, 16B per lane. Tracked by ASYNCcnt.
__device__ __forceinline__ void async_copy_b128(uint32_t lds_off, const void* gptr) {
  asm volatile("global_load_async_to_lds_b128 %0, %1, off"
               :: "v"(lds_off), "v"((unsigned long long)(uintptr_t)gptr)
               : "memory");
}
__device__ __forceinline__ void async_fence() {
  asm volatile("s_wait_asynccnt 0x0" ::: "memory");
}

// Load one 16x32 A/B fragment row for this lane.
// rowPtr points at element (row_for_this_lane, k=0); fragment layout:
//   elems[0..7]  = k = half*8 + 0..7
//   elems[8..15] = k = 16 + half*8 + 0..7
__device__ __forceinline__ v16bf load_frag(const __bf16* rowPtr, int half) {
  Frag16 f;
  f.q[0] = *(const uint4*)(rowPtr + half * 8);
  f.q[1] = *(const uint4*)(rowPtr + 16 + half * 8);
  return f.v;
}

__device__ __forceinline__ float hmax16(float x) {
#pragma unroll
  for (int m = 1; m < 16; m <<= 1) x = fmaxf(x, __shfl_xor(x, m, 32));
  return x;
}
__device__ __forceinline__ float hsum16(float x) {
#pragma unroll
  for (int m = 1; m < 16; m <<= 1) x += __shfl_xor(x, m, 32);
  return x;
}

// ---------------------------------------------------------------------------
// 0) f32 -> bf16 cast
// ---------------------------------------------------------------------------
__global__ void k_cast_bf16(const float* __restrict__ in,
                            __bf16* __restrict__ out, int n) {
  int i = blockIdx.x * blockDim.x + threadIdx.x;
  if (i < n) out[i] = (__bf16)in[i];
}

// ---------------------------------------------------------------------------
// 1) QKV GEMM: C[t][n] = sum_k H[t][k] * W[n][k] + bias[n]
//    Double-buffered async-LDS staging; 64x64 tile, 4 waves, 2x2 WMMA each.
//    Scatter n -> (head, {q,k,v}, d) into bf16 Q/K/V buffers [T][H*HDIM].
// ---------------------------------------------------------------------------
__global__ __launch_bounds__(128) void k_qkv_gemm(
    const __bf16* __restrict__ A, const __bf16* __restrict__ W,
    const float* __restrict__ bias, __bf16* __restrict__ Qo,
    __bf16* __restrict__ Ko, __bf16* __restrict__ Vo, int T) {
  __shared__ __bf16 lA[2][64 * 32];
  __shared__ __bf16 lB[2][64 * 32];

  const int t0 = blockIdx.x * 64;
  const int n0 = blockIdx.y * 64;
  const int tid = threadIdx.x;
  const int lane = tid & 31, wave = tid >> 5;
  const int ln16 = lane & 15, half = lane >> 4;
  const int mb = (wave >> 1) * 32;   // wave row offset in tile
  const int nb = (wave & 1) * 32;    // wave col offset in tile

  v8f acc[2][2] = {{{}, {}}, {{}, {}}};

  const int lr = tid >> 1;            // 0..63: row within tile
  const int lc = (tid & 1) * 16;      // 0 or 16: 16 elems = 32B per thread

  auto stage = [&](int buf, int kk) {
    const __bf16* asrc = A + (size_t)(t0 + lr) * EMBED + kk + lc;
    async_copy_b128(lds_addr32(&lA[buf][lr * 32 + lc]),     asrc);
    async_copy_b128(lds_addr32(&lA[buf][lr * 32 + lc + 8]), asrc + 8);
    const __bf16* wsrc = W + (size_t)(n0 + lr) * EMBED + kk + lc;
    async_copy_b128(lds_addr32(&lB[buf][lr * 32 + lc]),     wsrc);
    async_copy_b128(lds_addr32(&lB[buf][lr * 32 + lc + 8]), wsrc + 8);
  };

  stage(0, 0);
  async_fence();
  __syncthreads();

  for (int kk = 0; kk < EMBED; kk += 32) {
    const int buf = (kk >> 5) & 1;
    if (kk + 32 < EMBED) stage(buf ^ 1, kk + 32);  // DMA next tile under WMMA

    v16bf a0 = load_frag(&lA[buf][(mb + ln16) * 32], half);
    v16bf a1 = load_frag(&lA[buf][(mb + 16 + ln16) * 32], half);
    v16bf b0 = load_frag(&lB[buf][(nb + ln16) * 32], half);
    v16bf b1 = load_frag(&lB[buf][(nb + 16 + ln16) * 32], half);

    acc[0][0] = wmma_bf16(a0, b0, acc[0][0]);
    acc[0][1] = wmma_bf16(a0, b1, acc[0][1]);
    acc[1][0] = wmma_bf16(a1, b0, acc[1][0]);
    acc[1][1] = wmma_bf16(a1, b1, acc[1][1]);

    async_fence();
    __syncthreads();
  }

#pragma unroll
  for (int mt = 0; mt < 2; ++mt)
#pragma unroll
    for (int nt = 0; nt < 2; ++nt)
#pragma unroll
      for (int r = 0; r < 8; ++r) {
        int row = t0 + mb + mt * 16 + r + 8 * half;
        int col = n0 + nb + nt * 16 + ln16;
        float v = acc[mt][nt][r] + bias[col];
        // qkv reshape: col = h*192 + sel*64 + d
        int h = col / 192, rem = col % 192;
        int sel = rem >> 6, d = rem & 63;
        __bf16* dst = (sel == 0) ? Qo : (sel == 1) ? Ko : Vo;
        if (row < T) dst[(size_t)row * EMBED + h * HDIM + d] = (__bf16)v;
      }
}

// ---------------------------------------------------------------------------
// 2) Flash attention per (64-query tile, head). 4 waves; wave owns 16 q rows.
//    K tile staged via async-LDS DMA; V transposed through VGPRs; online
//    softmax in f32; QK^T and P·V via bf16 WMMA.
// ---------------------------------------------------------------------------
__global__ __launch_bounds__(128) void k_attn(
    const __bf16* __restrict__ Q, const __bf16* __restrict__ K,
    const __bf16* __restrict__ V, __bf16* __restrict__ CTX,
    const int* __restrict__ cu, int nseq, int T) {
  __shared__ __bf16 ldsK[32 * 64];       // [key][d]   (B-source for S)
  __shared__ __bf16 ldsVt[64 * 32];      // [d][key]   (B-source for P·V)
  __shared__ __bf16 ldsP[4 * 16 * 32];   // per-wave P strip [16 q][32 key]

  const int h = blockIdx.y;
  const int q0 = blockIdx.x * 64;
  const int tid = threadIdx.x;
  const int lane = tid & 31, wave = tid >> 5;
  const int ln16 = lane & 15, half = lane >> 4;
  const float scale = 0.125f;  // 1/sqrt(64)

  // find sequence containing q0
  int b = 0;
#pragma unroll 1
  for (; b < nseq - 1; ++b)
    if (q0 < cu[b + 1]) break;
  const int s1 = cu[b + 1];
  const int s0 = cu[b];

  // Q fragments for this wave's 16 rows, held in registers the whole time
  const int qrow = q0 + wave * 16 + ln16;
  const __bf16* qp = Q + (size_t)qrow * EMBED + h * HDIM;
  v16bf qa0 = load_frag(qp, half);       // d = 0..31
  v16bf qa1 = load_frag(qp + 32, half);  // d = 32..63

  float m_i[8], l_i[8];
#pragma unroll
  for (int r = 0; r < 8; ++r) { m_i[r] = -1e30f; l_i[r] = 0.0f; }
  v8f acc0 = {}, acc1 = {}, acc2 = {}, acc3 = {};

  __bf16* Pw = ldsP + wave * (16 * 32);

#pragma unroll 1
  for (int kb = s0; kb < s1; kb += 32) {
    __syncthreads();  // previous iteration's LDS reads complete
    {  // stage K tile [32 key][64 d] via async DMA (32B per thread)
      int kr = tid >> 2, kc = (tid & 3) * 16;
      int kg = kb + kr;
      if (kg < s1) {
        const __bf16* src = K + (size_t)kg * EMBED + h * HDIM + kc;
        async_copy_b128(lds_addr32(&ldsK[kr * 64 + kc]),     src);
        async_copy_b128(lds_addr32(&ldsK[kr * 64 + kc + 8]), src + 8);
        // lanes past seq end leave stale LDS; those keys are masked and their
        // P columns are exactly zero, so stale K/V never contributes.
      }
    }
    {  // stage V tile transposed: ldsVt[d][key]
      int key = tid & 31, dbase = (tid >> 5) * 16;
      int kg = kb + key;
      const __bf16* src = V + (size_t)kg * EMBED + h * HDIM + dbase;
      __align__(16) __bf16 tmp[16];
      *(uint4*)tmp       = (kg < s1) ? *(const uint4*)src       : make_uint4(0, 0, 0, 0);
      *(uint4*)(tmp + 8) = (kg < s1) ? *(const uint4*)(src + 8) : make_uint4(0, 0, 0, 0);
#pragma unroll
      for (int j = 0; j < 16; ++j) ldsVt[(dbase + j) * 32 + key] = tmp[j];
    }
    async_fence();
    __syncthreads();

    // S = Q * K^T (16 q x 32 keys per wave)
    v8f sA = {}, sB = {};
    {
      const __bf16* kp0 = ldsK + ln16 * 64;         // keys 0..15
      const __bf16* kp1 = ldsK + (16 + ln16) * 64;  // keys 16..31
      v16bf k0a = load_frag(kp0, half);
      v16bf k0b = load_frag(kp0 + 32, half);
      v16bf k1a = load_frag(kp1, half);
      v16bf k1b = load_frag(kp1 + 32, half);
      sA = wmma_bf16(qa0, k0a, sA);
      sA = wmma_bf16(qa1, k0b, sA);
      sB = wmma_bf16(qa0, k1a, sB);
      sB = wmma_bf16(qa1, k1b, sB);
    }

    // online softmax update; write P (bf16) into per-wave LDS strip
    const int key0 = kb + ln16, key1 = kb + 16 + ln16;
    const bool v0 = key0 < s1, v1 = key1 < s1;
#pragma unroll
    for (int r = 0; r < 8; ++r) {
      float a = v0 ? sA[r] * scale : -1e30f;
      float c = v1 ? sB[r] * scale : -1e30f;
      float mx = hmax16(fmaxf(a, c));
      float mnew = fmaxf(m_i[r], mx);
      float alpha = __expf(m_i[r] - mnew);
      float p0 = v0 ? __expf(a - mnew) : 0.0f;
      float p1 = v1 ? __expf(c - mnew) : 0.0f;
      l_i[r] = l_i[r] * alpha + hsum16(p0 + p1);
      m_i[r] = mnew;
      acc0[r] *= alpha; acc1[r] *= alpha; acc2[r] *= alpha; acc3[r] *= alpha;
      int rowm = r + 8 * half;  // C-layout row for this lane
      Pw[rowm * 32 + ln16]      = (__bf16)p0;
      Pw[rowm * 32 + 16 + ln16] = (__bf16)p1;
    }

    // ctx += P * V   (A = P 16x32 from LDS, B = V^T columns from LDS)
    v16bf pa  = load_frag(Pw + ln16 * 32, half);
    v16bf vb0 = load_frag(ldsVt + (0 * 16 + ln16) * 32, half);
    v16bf vb1 = load_frag(ldsVt + (1 * 16 + ln16) * 32, half);
    v16bf vb2 = load_frag(ldsVt + (2 * 16 + ln16) * 32, half);
    v16bf vb3 = load_frag(ldsVt + (3 * 16 + ln16) * 32, half);
    acc0 = wmma_bf16(pa, vb0, acc0);
    acc1 = wmma_bf16(pa, vb1, acc1);
    acc2 = wmma_bf16(pa, vb2, acc2);
    acc3 = wmma_bf16(pa, vb3, acc3);
  }

  // normalize and store context (bf16, [T][H*HDIM] layout)
#pragma unroll
  for (int r = 0; r < 8; ++r) {
    float inv = (l_i[r] > 0.0f) ? (1.0f / l_i[r]) : 0.0f;
    int row = q0 + wave * 16 + r + 8 * half;
    if (row < s1) {
      __bf16* dst = CTX + (size_t)row * EMBED + h * HDIM + ln16;
      dst[0]  = (__bf16)(acc0[r] * inv);
      dst[16] = (__bf16)(acc1[r] * inv);
      dst[32] = (__bf16)(acc2[r] * inv);
      dst[48] = (__bf16)(acc3[r] * inv);
    }
  }
}

// ---------------------------------------------------------------------------
// 3) Output GEMM: out[t][n] = sum_k CTX[t][k] * Wout[n][k] + bias[n]  (f32 out)
//    Same double-buffered async-LDS staging as the QKV GEMM.
// ---------------------------------------------------------------------------
__global__ __launch_bounds__(128) void k_out_gemm(
    const __bf16* __restrict__ A, const __bf16* __restrict__ W,
    const float* __restrict__ bias, float* __restrict__ out, int T) {
  __shared__ __bf16 lA[2][64 * 32];
  __shared__ __bf16 lB[2][64 * 32];

  const int t0 = blockIdx.x * 64;
  const int n0 = blockIdx.y * 64;
  const int tid = threadIdx.x;
  const int lane = tid & 31, wave = tid >> 5;
  const int ln16 = lane & 15, half = lane >> 4;
  const int mb = (wave >> 1) * 32;
  const int nb = (wave & 1) * 32;

  v8f acc[2][2] = {{{}, {}}, {{}, {}}};
  const int lr = tid >> 1;
  const int lc = (tid & 1) * 16;

  auto stage = [&](int buf, int kk) {
    const __bf16* asrc = A + (size_t)(t0 + lr) * EMBED + kk + lc;
    async_copy_b128(lds_addr32(&lA[buf][lr * 32 + lc]),     asrc);
    async_copy_b128(lds_addr32(&lA[buf][lr * 32 + lc + 8]), asrc + 8);
    const __bf16* wsrc = W + (size_t)(n0 + lr) * EMBED + kk + lc;
    async_copy_b128(lds_addr32(&lB[buf][lr * 32 + lc]),     wsrc);
    async_copy_b128(lds_addr32(&lB[buf][lr * 32 + lc + 8]), wsrc + 8);
  };

  stage(0, 0);
  async_fence();
  __syncthreads();

  for (int kk = 0; kk < EMBED; kk += 32) {
    const int buf = (kk >> 5) & 1;
    if (kk + 32 < EMBED) stage(buf ^ 1, kk + 32);

    v16bf a0 = load_frag(&lA[buf][(mb + ln16) * 32], half);
    v16bf a1 = load_frag(&lA[buf][(mb + 16 + ln16) * 32], half);
    v16bf b0 = load_frag(&lB[buf][(nb + ln16) * 32], half);
    v16bf b1 = load_frag(&lB[buf][(nb + 16 + ln16) * 32], half);

    acc[0][0] = wmma_bf16(a0, b0, acc[0][0]);
    acc[0][1] = wmma_bf16(a0, b1, acc[0][1]);
    acc[1][0] = wmma_bf16(a1, b0, acc[1][0]);
    acc[1][1] = wmma_bf16(a1, b1, acc[1][1]);

    async_fence();
    __syncthreads();
  }

#pragma unroll
  for (int mt = 0; mt < 2; ++mt)
#pragma unroll
    for (int nt = 0; nt < 2; ++nt)
#pragma unroll
      for (int r = 0; r < 8; ++r) {
        int row = t0 + mb + mt * 16 + r + 8 * half;
        int col = n0 + nb + nt * 16 + ln16;
        if (row < T) out[(size_t)row * EMBED + col] = acc[mt][nt][r] + bias[col];
      }
}

// ---------------------------------------------------------------------------
// Launch
// ---------------------------------------------------------------------------
extern "C" void kernel_launch(void* const* d_in, const int* in_sizes, int n_in,
                              void* d_out, int out_size, void* d_ws, size_t ws_size,
                              hipStream_t stream) {
  const float* hidden = (const float*)d_in[0];
  const float* proj_w = (const float*)d_in[1];
  const float* proj_b = (const float*)d_in[2];
  const float* out_w  = (const float*)d_in[3];
  const float* out_b  = (const float*)d_in[4];
  const int*   cu     = (const int*)d_in[5];

  const int T = in_sizes[0] / EMBED;       // 6144
  const int nseq = in_sizes[5] - 1;        // 8
  float* out = (float*)d_out;

  // workspace layout (bf16 buffers), 256B aligned
  size_t off = 0;
  auto alloc = [&](size_t bytes) { size_t c = off; off += (bytes + 255) & ~(size_t)255; return c; };
  char* ws = (char*)d_ws;
  __bf16* hb   = (__bf16*)(ws + alloc((size_t)T * EMBED * 2));
  __bf16* wqkv = (__bf16*)(ws + alloc((size_t)QKVDIM * EMBED * 2));
  __bf16* wout = (__bf16*)(ws + alloc((size_t)EMBED * EMBED * 2));
  __bf16* Qb   = (__bf16*)(ws + alloc((size_t)T * EMBED * 2));
  __bf16* Kb   = (__bf16*)(ws + alloc((size_t)T * EMBED * 2));
  __bf16* Vb   = (__bf16*)(ws + alloc((size_t)T * EMBED * 2));
  __bf16* ctx  = (__bf16*)(ws + alloc((size_t)T * EMBED * 2));

  // 0) casts
  {
    int n = T * EMBED;
    k_cast_bf16<<<(n + 255) / 256, 256, 0, stream>>>(hidden, hb, n);
    n = QKVDIM * EMBED;
    k_cast_bf16<<<(n + 255) / 256, 256, 0, stream>>>(proj_w, wqkv, n);
    n = EMBED * EMBED;
    k_cast_bf16<<<(n + 255) / 256, 256, 0, stream>>>(out_w, wout, n);
  }

  // 1) QKV projection
  {
    dim3 grid(T / 64, QKVDIM / 64);
    k_qkv_gemm<<<grid, 128, 0, stream>>>(hb, wqkv, proj_b, Qb, Kb, Vb, T);
  }

  // 2) attention (zero ctx first: deterministic even for rows outside tiles)
  hipMemsetAsync(ctx, 0, (size_t)T * EMBED * 2, stream);
  {
    dim3 grid(T / 64, NHEADS);
    k_attn<<<grid, 128, 0, stream>>>(Qb, Kb, Vb, ctx, cu, nseq, T);
  }

  // 3) output projection
  {
    dim3 grid(T / 64, EMBED / 64);
    k_out_gemm<<<grid, 128, 0, stream>>>(ctx, wout, out_b, out, T);
  }
}
